// Attention_17506286698899
// MI455X (gfx1250) — matrix operand order repaired and checked by reading, not compiled, not run on hardware
//
#include <hip/hip_runtime.h>
#include <hip/hip_bf16.h>
#include <math.h>

typedef __attribute__((ext_vector_type(16))) __bf16 v16bf;
typedef __attribute__((ext_vector_type(8)))  float  v8f;

#define HID   512
#define TWOH  1024
#define SEQ   2048
#define BATCH 64

// ---------------------------------------------------------------------------
// Kernel 0a: dec_feat[b][h] = decoder_state[b] @ W_s + b_s   (tiny GEMM)
// ---------------------------------------------------------------------------
__global__ __launch_bounds__(256) void dec_feat_kernel(
    const float* __restrict__ dec_state,  // [B][H]
    const float* __restrict__ W_s,        // [H][H]
    const float* __restrict__ b_s,        // [H]
    float* __restrict__ dec_feat)         // [B][H]
{
  __shared__ float srow[HID];
  int idx = blockIdx.x * blockDim.x + threadIdx.x;   // 0..B*H-1
  int b = idx / HID;
  int h = idx % HID;
  for (int k = threadIdx.x; k < HID; k += blockDim.x)
    srow[k] = dec_state[b * HID + k];
  __syncthreads();
  float acc = b_s[h];
  for (int k = 0; k < HID; ++k)
    acc += srow[k] * W_s[k * HID + h];   // coalesced over h
  dec_feat[idx] = acc;
}

// ---------------------------------------------------------------------------
// Kernel 0b: WhT[n][k] = bf16(W_h[k][n])  (transpose+convert, one-time 1 MB)
// ---------------------------------------------------------------------------
__global__ __launch_bounds__(256) void wh_transpose_kernel(
    const float* __restrict__ W_h,   // [1024 k][512 n]
    __bf16* __restrict__ WhT)        // [512 n][1024 k]
{
  __shared__ float tile[32][33];
  int n0 = blockIdx.x * 32;   // 16 blocks over n
  int k0 = blockIdx.y * 32;   // 32 blocks over k
  int tx = threadIdx.x;       // 0..31
  int ty = threadIdx.y;       // 0..7
#pragma unroll
  for (int i = 0; i < 4; ++i) {
    int k = k0 + ty + i * 8;
    tile[ty + i * 8][tx] = W_h[(size_t)k * HID + n0 + tx];  // coalesced over n
  }
  __syncthreads();
#pragma unroll
  for (int i = 0; i < 4; ++i) {
    int n = n0 + ty + i * 8;
    WhT[(size_t)n * TWOH + k0 + tx] = (__bf16)tile[tx][ty + i * 8]; // contiguous over k
  }
}

// ---------------------------------------------------------------------------
// A-fragment loader: 16x32 bf16 fragment, contiguous k order (same order as the B loads).
// lanes 0-15  (hi=0): halfs 0-15 = K kk+0..15
// lanes 16-31 (hi=1): halfs 0-15 = K kk+16..31
// ---------------------------------------------------------------------------
__device__ __forceinline__ v16bf load_a_frag(const float* __restrict__ arow, int kA)
{
  float4 x0 = *(const float4*)(arow + kA);
  float4 x1 = *(const float4*)(arow + kA + 4);
  float4 x2 = *(const float4*)(arow + kA + 8);
  float4 x3 = *(const float4*)(arow + kA + 12);
  v16bf a;
  a[0]=(__bf16)x0.x; a[1]=(__bf16)x0.y; a[2]=(__bf16)x0.z; a[3]=(__bf16)x0.w;
  a[4]=(__bf16)x1.x; a[5]=(__bf16)x1.y; a[6]=(__bf16)x1.z; a[7]=(__bf16)x1.w;
  a[8]=(__bf16)x2.x; a[9]=(__bf16)x2.y; a[10]=(__bf16)x2.z; a[11]=(__bf16)x2.w;
  a[12]=(__bf16)x3.x; a[13]=(__bf16)x3.y; a[14]=(__bf16)x3.z; a[15]=(__bf16)x3.w;
  return a;
}

// ---------------------------------------------------------------------------
// Kernel 1: fused  e[b][s] = v . tanh( enc[b][s] @ W_h + dec_feat[b] + cov*W_c )
// WG = 4 waves (128 thr); WG covers 32 S-rows, each wave owns all 32 rows
// (2 M-tiles) x 128 N columns (8 WMMA tile columns), so each B fragment
// feeds 2 WMMAs.  K = 1024, stepped by 32.
// ---------------------------------------------------------------------------
__global__ __launch_bounds__(128) void attn_scores_kernel(
    const float*  __restrict__ enc,      // [B][S][1024]
    const __bf16* __restrict__ WhT,      // [512][1024] bf16
    const float*  __restrict__ dec_feat, // [B][512]
    const float*  __restrict__ coverage, // [B][S]
    const float*  __restrict__ W_c,      // [512]
    const float*  __restrict__ vvec,     // [512]
    float* __restrict__ e_out)           // [B][S]
{
  const int wave  = threadIdx.x >> 5;
  const int lane  = threadIdx.x & 31;
  const int laneN = lane & 15;
  const int hi    = lane >> 4;              // 0: lanes 0-15, 1: lanes 16-31

  const int b  = blockIdx.x / (SEQ / 32);
  const int s0 = (blockIdx.x % (SEQ / 32)) * 32;
  const int nb = wave * 128;

  const float* arow0 = enc + ((size_t)b * SEQ + (s0 + laneN)) * TWOH;       // rows s0..s0+15
  const float* arow1 = arow0 + (size_t)16 * TWOH;                           // rows s0+16..s0+31

  v8f acc[2][8] = {};

  for (int kk = 0; kk < TWOH; kk += 32) {
    const int kA = kk + hi * 16;
    const v16bf a0 = load_a_frag(arow0, kA);
    const v16bf a1 = load_a_frag(arow1, kA);
    __builtin_prefetch(arow0 + kk + 64, 0, 1);
    __builtin_prefetch(arow1 + kk + 64, 0, 1);

#pragma unroll
    for (int t = 0; t < 8; ++t) {
      const int n = nb + t * 16 + laneN;
      const v16bf brow = *(const v16bf*)(WhT + (size_t)n * TWOH + kk + hi * 16);
      acc[0][t] = __builtin_amdgcn_wmma_f32_16x16x32_bf16(
          false, a0, false, brow, (short)0, acc[0][t], false, false);
      acc[1][t] = __builtin_amdgcn_wmma_f32_16x16x32_bf16(
          false, a1, false, brow, (short)0, acc[1][t], false, false);
    }
  }

  // ---- epilogue: bias + coverage feature, tanh, dot with v (partial over N)
  float cm[2][8];
#pragma unroll
  for (int u = 0; u < 2; ++u)
#pragma unroll
    for (int r = 0; r < 8; ++r)
      cm[u][r] = coverage[(size_t)b * SEQ + s0 + u * 16 + r + hi * 8];

  float pe[2][8];
#pragma unroll
  for (int u = 0; u < 2; ++u)
#pragma unroll
    for (int r = 0; r < 8; ++r) pe[u][r] = 0.0f;

#pragma unroll
  for (int t = 0; t < 8; ++t) {
    const int n = nb + t * 16 + laneN;
    const float df = dec_feat[b * HID + n];
    const float wc = W_c[n];
    const float vv = vvec[n];
#pragma unroll
    for (int u = 0; u < 2; ++u)
#pragma unroll
      for (int r = 0; r < 8; ++r) {
        float x = acc[u][t][r] + df + cm[u][r] * wc;
        pe[u][r] += vv * tanhf(x);
      }
  }

  // reduce over the 16 lanes of each half (N reduction within wave)
#pragma unroll
  for (int u = 0; u < 2; ++u)
#pragma unroll
    for (int r = 0; r < 8; ++r) {
      pe[u][r] += __shfl_xor(pe[u][r], 1);
      pe[u][r] += __shfl_xor(pe[u][r], 2);
      pe[u][r] += __shfl_xor(pe[u][r], 4);
      pe[u][r] += __shfl_xor(pe[u][r], 8);
    }

  // combine the 4 waves' partial sums per row via LDS
  __shared__ float eacc[32];
  if (threadIdx.x < 32) eacc[threadIdx.x] = 0.0f;
  __syncthreads();
  if (laneN == 0) {
#pragma unroll
    for (int u = 0; u < 2; ++u)
#pragma unroll
      for (int r = 0; r < 8; ++r)
        atomicAdd(&eacc[u * 16 + r + hi * 8], pe[u][r]);
  }
  __syncthreads();
  if (threadIdx.x < 32)
    e_out[(size_t)b * SEQ + s0 + threadIdx.x] = eacc[threadIdx.x];
}

// ---------------------------------------------------------------------------
// Kernel 2: masked softmax over S + new_coverage
// ---------------------------------------------------------------------------
__global__ __launch_bounds__(256) void softmax_cov_kernel(
    const float* __restrict__ e,
    const int*   __restrict__ mask,
    const float* __restrict__ coverage,
    float* __restrict__ attn,
    float* __restrict__ newcov)
{
  __shared__ float red[256];
  const int b = blockIdx.x;
  const int tid = threadIdx.x;
  float ev[8];
  float lmax = -3.4e38f;
#pragma unroll
  for (int i = 0; i < 8; ++i) {
    int s = tid + i * 256;
    float x = e[(size_t)b * SEQ + s];
    if (mask[(size_t)b * SEQ + s] == 0) x = -10000.0f;
    ev[i] = x;
    lmax = fmaxf(lmax, x);
  }
  red[tid] = lmax; __syncthreads();
  for (int off = 128; off > 0; off >>= 1) {
    if (tid < off) red[tid] = fmaxf(red[tid], red[tid + off]);
    __syncthreads();
  }
  const float m = red[0];
  __syncthreads();

  float lsum = 0.0f;
#pragma unroll
  for (int i = 0; i < 8; ++i) { ev[i] = __expf(ev[i] - m); lsum += ev[i]; }
  red[tid] = lsum; __syncthreads();
  for (int off = 128; off > 0; off >>= 1) {
    if (tid < off) red[tid] += red[tid + off];
    __syncthreads();
  }
  const float inv = 1.0f / red[0];
#pragma unroll
  for (int i = 0; i < 8; ++i) {
    int s = tid + i * 256;
    float p = ev[i] * inv;
    attn[(size_t)b * SEQ + s] = p;
    newcov[(size_t)b * SEQ + s] = coverage[(size_t)b * SEQ + s] + p;
  }
}

// ---------------------------------------------------------------------------
// Kernel 3: context[b] = sum_s attn[b][s] * enc[b][s][:]   (2nd 512 MB stream)
// ---------------------------------------------------------------------------
__global__ __launch_bounds__(256) void zero_kernel(float* __restrict__ p, int n)
{
  int i = blockIdx.x * blockDim.x + threadIdx.x;
  if (i < n) p[i] = 0.0f;
}

__global__ __launch_bounds__(256) void context_kernel(
    const float* __restrict__ enc,   // [B][S][1024]
    const float* __restrict__ attn,  // [B][S]
    float* __restrict__ ctx)         // [B][1024]
{
  __shared__ float pa[128];
  const int b  = blockIdx.x >> 4;
  const int sc = blockIdx.x & 15;
  const int s0 = sc * 128;
  const int tid = threadIdx.x;
  if (tid < 128) pa[tid] = attn[(size_t)b * SEQ + s0 + tid];
  __syncthreads();

  float4 c; c.x = 0.f; c.y = 0.f; c.z = 0.f; c.w = 0.f;
  // each thread owns 4 consecutive columns -> one b128 load per row
  const float4* base = (const float4*)(enc + ((size_t)b * SEQ + s0) * TWOH);
  for (int s = 0; s < 128; ++s) {
    const float4* row = base + (size_t)s * (TWOH / 4);
    __builtin_prefetch(row + (TWOH / 4), 0, 0);   // next row
    const float4 q = row[tid];
    const float p = pa[s];
    c.x += p * q.x;
    c.y += p * q.y;
    c.z += p * q.z;
    c.w += p * q.w;
  }
  float* dst = &ctx[(size_t)b * TWOH + tid * 4];
  atomicAdd(dst + 0, c.x);
  atomicAdd(dst + 1, c.y);
  atomicAdd(dst + 2, c.z);
  atomicAdd(dst + 3, c.w);
}

// ---------------------------------------------------------------------------
extern "C" void kernel_launch(void* const* d_in, const int* in_sizes, int n_in,
                              void* d_out, int out_size, void* d_ws, size_t ws_size,
                              hipStream_t stream) {
  (void)in_sizes; (void)n_in; (void)out_size; (void)ws_size;

  const float* dec_state = (const float*)d_in[0];   // (64,512)
  const float* enc       = (const float*)d_in[1];   // (64,2048,1024)
  const int*   mask      = (const int*)  d_in[2];   // (64,2048)
  const float* coverage  = (const float*)d_in[3];   // (64,2048)
  const float* W_h       = (const float*)d_in[4];   // (1024,512)
  const float* W_s       = (const float*)d_in[5];   // (512,512)
  const float* b_s       = (const float*)d_in[6];   // (512,)
  const float* W_c       = (const float*)d_in[7];   // (1,512)
  const float* v         = (const float*)d_in[8];   // (512,1)

  float* out  = (float*)d_out;
  float* ctx  = out;                         // 64*1024
  float* attn = out + BATCH * TWOH;          // 64*2048
  float* ncov = attn + BATCH * SEQ;          // 64*2048

  char* ws = (char*)d_ws;
  __bf16* WhT  = (__bf16*)ws;                                  // 1 MB
  float*  dfeat = (float*)(ws + (1u << 20));                   // 128 KB
  float*  evals = (float*)(ws + (1u << 20) + (128u << 10));    // 512 KB

  dec_feat_kernel<<<BATCH * HID / 256, 256, 0, stream>>>(dec_state, W_s, b_s, dfeat);
  wh_transpose_kernel<<<dim3(HID / 32, TWOH / 32), dim3(32, 8), 0, stream>>>(W_h, WhT);
  attn_scores_kernel<<<BATCH * (SEQ / 32), 128, 0, stream>>>(
      enc, WhT, dfeat, coverage, W_c, v, evals);
  softmax_cov_kernel<<<BATCH, 256, 0, stream>>>(evals, mask, coverage, attn, ncov);
  zero_kernel<<<(BATCH * TWOH + 255) / 256, 256, 0, stream>>>(ctx, BATCH * TWOH);
  context_kernel<<<BATCH * 16, 256, 0, stream>>>(enc, attn, ctx);
}
